// WindowSelfAttentionBlock_25718264168553
// MI455X (gfx1250) — matrix operand order repaired and checked by reading, not compiled, not run on hardware
//
#include <hip/hip_runtime.h>

// ---------------------------------------------------------------------------
// Types for gfx1250 WMMA (wave32)
// ---------------------------------------------------------------------------
typedef __bf16 bf16_t;
typedef __attribute__((ext_vector_type(16))) __bf16 v16bf;
typedef __attribute__((ext_vector_type(8)))  __bf16 v8bf;
typedef __attribute__((ext_vector_type(8)))  float  v8f;

__device__ __forceinline__ v8f wmma_bf16(v16bf a, v16bf b, v8f c) {
    // v_wmma_f32_16x16x32_bf16: (neg_a, A, neg_b, B, c_mod, C, reuse_a, reuse_b)
    return __builtin_amdgcn_wmma_f32_16x16x32_bf16(false, a, false, b,
                                                   (short)0, c, false, false);
}

// load 16 bf16 as two 16B chunks (16B-aligned each); works for LDS & global
__device__ __forceinline__ v16bf ld16(const bf16_t* p) {
    v8bf lo = *(const v8bf*)p;
    v8bf hi = *(const v8bf*)(p + 8);
    v16bf r;
#pragma unroll
    for (int i = 0; i < 8; ++i) { r[i] = lo[i]; r[i + 8] = hi[i]; }
    return r;
}
// A-fragment split load: elements 0..7 at p, elements 8..15 at p+16
__device__ __forceinline__ v16bf ld16_split(const bf16_t* p) {
    v8bf lo = *(const v8bf*)p;
    v8bf hi = *(const v8bf*)(p + 16);
    v16bf r;
#pragma unroll
    for (int i = 0; i < 8; ++i) { r[i] = lo[i]; r[i + 8] = hi[i]; }
    return r;
}

// ---------------------------------------------------------------------------
// Static problem config
// ---------------------------------------------------------------------------
#define BB 32
#define CC 256
#define HH 64
#define WW 64
#define NH 8
#define DK 32
#define WS 8
#define LL 64
#define SCALE_F 0.17677669529663687f   // 32^-0.5

// LDS layout strides (padded against bank conflicts, keep 16B alignment)
#define XSTR   264   // x window, token-major, bf16 elems per token row
#define QKSTR  264   // q/k, token-major
#define ATTSTR 68    // attn, f32 elems per row (68*4 = 272B, 16B aligned)

#define SM_QK  (2 * 64 * QKSTR * 2)            // 67584 B  (q then k)
#define SM_ATT (8 * 64 * ATTSTR * 4)           // 139264 B (overlaps x staging)
#define SMEM1  (SM_QK + SM_ATT)                // 206848 B  (< 320KB/WGP)

// ---------------------------------------------------------------------------
// Kernel 0: pack qkv weights (first 512 rows: q,k only; v is dead code in the
// reference) and pointwise weights into WMMA A-fragment order, f32 -> bf16.
// Consumer reads: wpack[((mt*8+kt)*32+lane)*16 + i] contiguous per lane.
// ---------------------------------------------------------------------------
__global__ __launch_bounds__(256) void pack_w_kernel(
    const float* __restrict__ qkvw, const float* __restrict__ dpw,
    bf16_t* __restrict__ wqk, bf16_t* __restrict__ wdp)
{
    int id = blockIdx.x * 256 + threadIdx.x;           // 196608 total
    const float* src; bf16_t* dst; int lid;
    if (id < 512 * 256) { src = qkvw; dst = wqk; lid = id; }
    else                { src = dpw;  dst = wdp; lid = id - 512 * 256; }
    int i    = lid & 15;
    int lane = (lid >> 4) & 31;
    int kt   = (lid >> 9) & 7;
    int mt   = lid >> 12;
    int lrow = lane & 15, lhi = lane >> 4;
    int row  = mt * 16 + lrow;
    int k    = kt * 32 + ((i < 8) ? (lhi * 8 + i) : (16 + lhi * 8 + (i - 8)));
    dst[lid] = (bf16_t)src[row * 256 + k];
}

// ---------------------------------------------------------------------------
// Kernel 1: fused QK-GEMM + window attention. 1 block = 1 window (2048 blocks),
// 256 threads = 8 waves; wave h owns head h for the attention part.
// ---------------------------------------------------------------------------
__global__ __launch_bounds__(256) void win_attn_kernel(
    const float* __restrict__ x, const bf16_t* __restrict__ wqk,
    const float* __restrict__ qkv_b, const float* __restrict__ rpb,
    bf16_t* __restrict__ msa)
{
    extern __shared__ char smem[];
    bf16_t* qks = (bf16_t*)smem;                 // [2][64][QKSTR] q then k
    bf16_t* xs  = (bf16_t*)(smem + SM_QK);       // [64][XSTR] (phase A/B only)
    float*  att = (float*)(smem + SM_QK);        // [8][64][ATTSTR] (phase C)

    const int wid = blockIdx.x;
    const int b   = wid >> 6;
    const int wy  = (wid >> 3) & 7, wx = wid & 7;
    const int gh0 = wy * 8, gw0 = wx * 8;
    const int t    = threadIdx.x;
    const int lane = t & 31, wave = t >> 5;
    const int lrow = lane & 15, lhi = lane >> 4;

    // ---- Phase A: stage x window token-major bf16 into xs[token][c] ----
    {
        const int ly = t & 7, coff = t >> 3;     // coff in [0,32)
        for (int cb = 0; cb < 256; cb += 32) {
            int c = cb + coff;
            const float* px = x + (((size_t)(b * 256 + c) * 64 + gh0 + ly) * 64 + gw0);
            float4 f0 = *(const float4*)px;
            float4 f1 = *(const float4*)(px + 4);
            bf16_t* pw = xs + c;
            int tok = ly * 8;
            pw[(tok + 0) * XSTR] = (bf16_t)f0.x;
            pw[(tok + 1) * XSTR] = (bf16_t)f0.y;
            pw[(tok + 2) * XSTR] = (bf16_t)f0.z;
            pw[(tok + 3) * XSTR] = (bf16_t)f0.w;
            pw[(tok + 4) * XSTR] = (bf16_t)f1.x;
            pw[(tok + 5) * XSTR] = (bf16_t)f1.y;
            pw[(tok + 6) * XSTR] = (bf16_t)f1.z;
            pw[(tok + 7) * XSTR] = (bf16_t)f1.w;
        }
    }
    __syncthreads();

    // ---- Phase B: [q;k](512x64) = Wqk(512x256) * Xwin(256x64), WMMA ----
    for (int tile = wave; tile < 128; tile += 8) {
        int mt = tile >> 2, nt = tile & 3;
        v8f acc = {};
#pragma unroll
        for (int kt = 0; kt < 8; ++kt) {
            v16bf a  = ld16(wqk + ((size_t)((mt * 8 + kt) * 32 + lane)) * 16);
            v16bf bm = ld16_split(xs + (nt * 16 + lrow) * XSTR + kt * 32 + lhi * 8);
            acc = wmma_bf16(a, bm, acc);
        }
        int tok   = nt * 16 + lrow;
        int mbase = mt * 16 + lhi * 8;           // output channel base (0..504)
        int sel   = mbase >> 8;                  // 0 = q, 1 = k
        int mloc  = mbase & 255;
        v8bf vv;
#pragma unroll
        for (int r = 0; r < 8; ++r)
            vv[r] = (bf16_t)(acc[r] + qkv_b[mbase + r]);
        *(v8bf*)(qks + ((sel * 64 + tok) * QKSTR + mloc)) = vv;   // ds_store_b128
    }
    __syncthreads();

    // ---- Phase C: per-head attention (wave h -> head h) ----
    const int h = wave;
    float* atth = att + h * 64 * ATTSTR;
    const bf16_t* qs = qks;
    const bf16_t* ks = qks + 64 * QKSTR;

    // attn = (q^T k) * scale + relpos bias  (K = DK = 32 -> one WMMA per tile)
#pragma unroll
    for (int tt = 0; tt < 4; ++tt) {
#pragma unroll
        for (int ss = 0; ss < 4; ++ss) {
            v8f acc = {};
            v16bf a  = ld16_split(qs + (tt * 16 + lrow) * QKSTR + h * 32 + lhi * 8);
            v16bf bm = ld16_split(ks + (ss * 16 + lrow) * QKSTR + h * 32 + lhi * 8);
            acc = wmma_bf16(a, bm, acc);
#pragma unroll
            for (int r = 0; r < 8; ++r) {
                int ti = tt * 16 + 8 * lhi + r;
                int si = ss * 16 + lrow;
                int idx = ((ti >> 3) - (si >> 3) + 7) * 15 + ((ti & 7) - (si & 7) + 7);
                atth[ti * ATTSTR + si] = acc[r] * SCALE_F + rpb[idx * 8 + h];
            }
        }
    }

    // softmax over rows (wave-local; 2 rows per lane)
#pragma unroll
    for (int p = 0; p < 2; ++p) {
        int row = p * 32 + lane;
        float* pr = atth + row * ATTSTR;
        float rv[64];
        float mx = -3.0e38f;
#pragma unroll
        for (int s = 0; s < 64; ++s) { rv[s] = pr[s]; mx = fmaxf(mx, rv[s]); }
        float sum = 0.f;
#pragma unroll
        for (int s = 0; s < 64; ++s) { rv[s] = __expf(rv[s] - mx); sum += rv[s]; }
        float inv = 1.0f / sum;
#pragma unroll
        for (int s = 0; s < 64; ++s) pr[s] = rv[s] * inv;
    }

    // out(32x64) = V(32x64) * attn^T ; V = K (faithful to reference bug)
#pragma unroll
    for (int mt = 0; mt < 2; ++mt) {
#pragma unroll
        for (int nt = 0; nt < 4; ++nt) {
            v8f acc = {};
#pragma unroll
            for (int kt = 0; kt < 2; ++kt) {
                // A[c][s] = k[c][s] = ks[token s][h*32+c] (strided gather)
                const bf16_t* abase = ks + h * 32 + mt * 16 + lrow;
                v16bf a;
#pragma unroll
                for (int i = 0; i < 16; ++i) {
                    int s = kt * 32 + ((i < 8) ? (lhi * 8 + i)
                                               : (16 + lhi * 8 + (i - 8)));
                    a[i] = abase[s * QKSTR];
                }
                // B[s][t] = attn[t][s]: 8 f32 at lhi*8 and 8 at 16+lhi*8, convert to bf16
                const float4* pb = (const float4*)(atth + (nt * 16 + lrow) * ATTSTR
                                                   + kt * 32 + lhi * 8);
                float4 q0 = pb[0], q1 = pb[1], q2 = pb[4], q3 = pb[5];
                v16bf bm;
                bm[0]=(bf16_t)q0.x; bm[1]=(bf16_t)q0.y; bm[2]=(bf16_t)q0.z; bm[3]=(bf16_t)q0.w;
                bm[4]=(bf16_t)q1.x; bm[5]=(bf16_t)q1.y; bm[6]=(bf16_t)q1.z; bm[7]=(bf16_t)q1.w;
                bm[8]=(bf16_t)q2.x; bm[9]=(bf16_t)q2.y; bm[10]=(bf16_t)q2.z; bm[11]=(bf16_t)q2.w;
                bm[12]=(bf16_t)q3.x; bm[13]=(bf16_t)q3.y; bm[14]=(bf16_t)q3.z; bm[15]=(bf16_t)q3.w;
                acc = wmma_bf16(a, bm, acc);
            }
            // write window-reversed NCHW bf16
            int tok = nt * 16 + lrow;
            int gh = gh0 + (tok >> 3), gw = gw0 + (tok & 7);
            int cb = h * 32 + mt * 16 + 8 * lhi;
            bf16_t* pm = msa + ((size_t)(b * 256 + cb) * 4096) + gh * 64 + gw;
#pragma unroll
            for (int r = 0; r < 8; ++r) pm[(size_t)r * 4096] = (bf16_t)acc[r];
        }
    }
}

// ---------------------------------------------------------------------------
// Kernel 2: depthwise 3x3, pad 1 (memory-bound elementwise), bf16 -> bf16
// ---------------------------------------------------------------------------
__global__ __launch_bounds__(256) void dw_kernel(
    const bf16_t* __restrict__ msa, const float* __restrict__ spw,
    const float* __restrict__ spb, bf16_t* __restrict__ out)
{
    int idx = blockIdx.x * 256 + threadIdx.x;      // B*C*H*W threads
    int gw = idx & 63, gh = (idx >> 6) & 63, c = (idx >> 12) & 255, b = idx >> 20;
    const bf16_t* base = msa + (size_t)(b * 256 + c) * 4096;
    float acc = spb[c];
#pragma unroll
    for (int ky = 0; ky < 3; ++ky) {
        int y = gh + ky - 1;
        if ((unsigned)y < 64u) {
#pragma unroll
            for (int kx = 0; kx < 3; ++kx) {
                int xw = gw + kx - 1;
                if ((unsigned)xw < 64u)
                    acc += spw[c * 9 + ky * 3 + kx] * (float)base[y * 64 + xw];
            }
        }
    }
    out[idx] = (bf16_t)acc;
}

// ---------------------------------------------------------------------------
// Kernel 3: pointwise 1x1 GEMM out(256 x 131072) = Wdp(256x256) * dw, WMMA.
// 1 block = 128 pixels (within one image), 8 waves x 16 tiles, f32 output.
// ---------------------------------------------------------------------------
__global__ __launch_bounds__(256) void pw_kernel(
    const bf16_t* __restrict__ dw, const bf16_t* __restrict__ wdp,
    const float* __restrict__ dpb, float* __restrict__ out)
{
    extern __shared__ char smem[];
    bf16_t* ds = (bf16_t*)smem;                    // [128][264] pixel-major

    const size_t n0 = (size_t)blockIdx.x * 128;
    const int b   = (int)(n0 >> 12);
    const int hw0 = (int)(n0 & 4095);
    const int t    = threadIdx.x;
    const int lane = t & 31, wave = t >> 5;
    const int lrow = lane & 15, lhi = lane >> 4;

    // stage transposed tile: ds[pixel][ci] <- dw[b][ci][hw0+pixel]
    {
        const bf16_t* src = dw + ((size_t)b * 256 + t) * 4096 + hw0;
        for (int j0 = 0; j0 < 128; j0 += 16) {
            v16bf vv = ld16(src + j0);
#pragma unroll
            for (int j = 0; j < 16; ++j) ds[(j0 + j) * 264 + t] = vv[j];
        }
    }
    __syncthreads();

    for (int tile = wave; tile < 128; tile += 8) {
        int mt = tile >> 3, nt = tile & 7;
        v8f acc = {};
#pragma unroll
        for (int kt = 0; kt < 8; ++kt) {
            v16bf a  = ld16(wdp + ((size_t)((mt * 8 + kt) * 32 + lane)) * 16);
            v16bf bm = ld16_split(ds + (nt * 16 + lrow) * 264 + kt * 32 + lhi * 8);
            acc = wmma_bf16(a, bm, acc);
        }
        int j   = nt * 16 + lrow;
        int hw  = hw0 + j;
        int co0 = mt * 16 + 8 * lhi;
        float* po = out + ((size_t)b * 256 + co0) * 4096 + hw;
#pragma unroll
        for (int r = 0; r < 8; ++r)
            po[(size_t)r * 4096] = acc[r] + dpb[co0 + r];
    }
}

// ---------------------------------------------------------------------------
// Launch: inputs are x, qkv_w, qkv_b, rpb_table, sp_w, sp_b, dp_w, dp_b (f32)
// Workspace layout (bytes):
//   [0, 256K)         packed Wqk bf16 (512x256)
//   [256K, 384K)      packed Wdp bf16 (256x256)
//   [384K, +64MB)     msa bf16 (B,C,H,W)
//   [+64MB, +128MB)   dw  bf16 (B,C,H,W)          total ~128.4 MB
// ---------------------------------------------------------------------------
extern "C" void kernel_launch(void* const* d_in, const int* in_sizes, int n_in,
                              void* d_out, int out_size, void* d_ws, size_t ws_size,
                              hipStream_t stream) {
    (void)in_sizes; (void)n_in; (void)out_size; (void)ws_size;
    const float* x     = (const float*)d_in[0];
    const float* qkv_w = (const float*)d_in[1];
    const float* qkv_b = (const float*)d_in[2];
    const float* rpb   = (const float*)d_in[3];
    const float* sp_w  = (const float*)d_in[4];
    const float* sp_b  = (const float*)d_in[5];
    const float* dp_w  = (const float*)d_in[6];
    const float* dp_b  = (const float*)d_in[7];
    float* out = (float*)d_out;

    char* ws = (char*)d_ws;
    bf16_t* wqk = (bf16_t*)ws;
    bf16_t* wdp = (bf16_t*)(ws + 262144);
    bf16_t* msa = (bf16_t*)(ws + 393216);
    bf16_t* dwb = (bf16_t*)(ws + 393216 + 67108864);

    pack_w_kernel<<<768, 256, 0, stream>>>(qkv_w, dp_w, wqk, wdp);
    win_attn_kernel<<<2048, 256, SMEM1, stream>>>(x, wqk, qkv_b, rpb, msa);
    dw_kernel<<<(BB * CC * HH * WW) / 256, 256, 0, stream>>>(msa, sp_w, sp_b, dwb);
    pw_kernel<<<1024, 256, 128 * 264 * 2, stream>>>(dwb, wdp, dp_b, out);
}
